// ManifoldDynamic_52467320488098
// MI455X (gfx1250) — matrix operand order repaired and checked by reading, not compiled, not run on hardware
//
#include <hip/hip_runtime.h>
#include <math.h>

// ---------------------------------------------------------------------------
// ManifoldDynamic on MI455X (gfx1250, wave32, WMMA)
//
// Shapes: BS=128, N=256, H=1024.  All GEMMs are bf16 WMMA (16x16x32) with
// f32 accumulation.  The Jacobian is contracted analytically:
//   J[b] = Wk2 . diag(1 - tanh^2(pre_k[b])) . Wk1
//   Jf   = ((f_out @ Wk2) * s) @ Wk1      (same for JG with G_out)
// B tiles are staged global->LDS with CDNA5 async-to-LDS DMA (ASYNCcnt),
// double buffered, so the DMA overlaps the WMMA body instead of stalling
// on loadcnt at issue like a VGPR round-trip would.
// ---------------------------------------------------------------------------

typedef __attribute__((ext_vector_type(16))) __bf16 v16bf;
typedef __attribute__((ext_vector_type(8)))  __bf16 v8bf;
typedef __attribute__((ext_vector_type(8)))  float  v8f;

#define BS_DIM 128
#define N_DIM  256
#define H_DIM  1024

#define EPI_BIAS  0
#define EPI_TANH  1
#define EPI_SCALE 2

#define LDS_COL_STRIDE 40           // halfs (80 B) -> conflict-free 16-col reads
#define LDS_BUF_BYTES  (64 * LDS_COL_STRIDE * 2)   // 5120 B per buffer

__device__ __forceinline__ __bf16 f2bf(float f) {
  unsigned u = __float_as_uint(f);
  unsigned r = (u + 0x7FFFu + ((u >> 16) & 1u)) >> 16;
  unsigned short s = (unsigned short)r;
  return __builtin_bit_cast(__bf16, s);
}

union ABfrag { v16bf v; v8bf h[2]; };

// ---------------------------------------------------------------------------
// NT GEMM: C[128, Nc] = A[128, K] @ B[Nc, K]^T   (A, B row-major bf16, K contig)
// One 256-thread block (8 waves) covers all 128 rows and a 64-column stripe.
// B stripe staged through LDS with global_load_async_to_lds_b128 (double
// buffered, ASYNCcnt).  Each wave: 16x64 tile = 4 v8f accumulators ->
// 4 v_wmma_f32_16x16x32_bf16 per 32-wide K step.
// ---------------------------------------------------------------------------
template <int MODE>
__global__ __launch_bounds__(256)
void gemm_nt_bf16(const __bf16* __restrict__ A,
                  const __bf16* __restrict__ B,
                  const float*  __restrict__ bias,    // [Nc] or null
                  float*        __restrict__ outF,    // f32 out or null
                  __bf16*       __restrict__ outB,    // bf16 out or null
                  float*        __restrict__ sOut,    // 1 - tanh^2 capture or null
                  const float*  __restrict__ scaleIn, // [128, Nc] for EPI_SCALE
                  int Nc, int K)
{
  // 2 buffers x 64 columns x 32 K bf16, column stride 40 halfs (80B).
  __shared__ __align__(16) __bf16 ldsB[2][64 * LDS_COL_STRIDE];

  const int tid   = threadIdx.x;
  const int lane  = tid & 31;
  const int wave  = tid >> 5;
  const int nBase = blockIdx.x << 6;

  // A fragment addressing (16-bit A 16x32 layout):
  //   lanes 0-15: row = lane,   K chunks [0..7]  and [16..23]
  //   lanes16-31: row = lane-16, K chunks [8..15] and [24..31]
  const int mrow   = (wave << 4) + (lane & 15);
  const int khalfA = (lane >> 4) << 3;          // 0 or 8 (elements)

  // Cooperative B-tile staging: 256 threads x b128 = 64 cols x 32 K.
  const int lcol = tid >> 2;                    // 0..63
  const int lkc  = (tid & 3) << 3;              // K sub-chunk 0/8/16/24 (elems)
  // LDS byte offset of this thread's staging slot in buffer 0 (low 32 bits of
  // the generic shared pointer == LDS byte address; HW adds the wave LDS base).
  const unsigned ldsStoreOff =
      (unsigned)(size_t)&ldsB[0][lcol * LDS_COL_STRIDE + lkc];

  v8f acc[4];
#pragma unroll
  for (int s = 0; s < 4; ++s) acc[s] = (v8f){0, 0, 0, 0, 0, 0, 0, 0};

  const int nk = K >> 5;

  // Prologue: async-stage tile 0 into buffer 0.
  {
    const __bf16* gp = B + (size_t)(nBase + lcol) * K + lkc;
    asm volatile("global_load_async_to_lds_b128 %0, %1, off"
                 :: "v"(ldsStoreOff), "v"(gp) : "memory");
  }
  asm volatile("s_wait_asynccnt 0x0" ::: "memory");
  __syncthreads();

  int buf = 0;
  for (int kt = 0; kt < nk; ++kt) {
    const int k0 = kt << 5;

    // Kick the DMA for the next tile first; it overlaps the WMMA body below.
    if (kt + 1 < nk) {
      const __bf16* gp = B + (size_t)(nBase + lcol) * K + (k0 + 32) + lkc;
      const unsigned dst = ldsStoreOff + (unsigned)(buf ^ 1) * LDS_BUF_BYTES;
      asm volatile("global_load_async_to_lds_b128 %0, %1, off"
                   :: "v"(dst), "v"(gp) : "memory");
    }

    // A fragment: two b128 loads per lane.
    ABfrag a;
    const __bf16* arow = A + (size_t)mrow * K + k0 + khalfA;
    a.h[0] = *(const v8bf*)(arow);
    a.h[1] = *(const v8bf*)(arow + 16);

    // Four 16-col subtiles from LDS, reusing the A fragment.
#pragma unroll
    for (int s = 0; s < 4; ++s) {
      ABfrag bf;
      const int base = ((s << 4) + (lane & 15)) * LDS_COL_STRIDE + ((lane >> 4) << 3);
      bf.h[0] = *(const v8bf*)&ldsB[buf][base];
      bf.h[1] = *(const v8bf*)&ldsB[buf][base + 16];
      acc[s] = __builtin_amdgcn_wmma_f32_16x16x32_bf16(
          false, a.v, false, bf.v, (short)0, acc[s], false, false);
    }

    // All async DMA into the other buffer must land before anyone crosses.
    asm volatile("s_wait_asynccnt 0x0" ::: "memory");
    __syncthreads();
    buf ^= 1;
  }

  // Epilogue. C/D layout: VGPR r -> (M = r + 8*(lane>=16), N = lane&15).
  const int rowBase = (wave << 4) + ((lane >> 4) << 3);
  const int colLane = lane & 15;
#pragma unroll
  for (int s = 0; s < 4; ++s) {
    const int col = nBase + (s << 4) + colLane;
    const float bv = bias ? bias[col] : 0.0f;
#pragma unroll
    for (int r = 0; r < 8; ++r) {
      const int row = rowBase + r;
      const size_t idx = (size_t)row * Nc + col;
      float t = acc[s][r];
      if (MODE == EPI_TANH) {
        t = tanhf(t + bv);
        outB[idx] = f2bf(t);
        if (sOut) sOut[idx] = 1.0f - t * t;     // sech^2 for the Jacobian
      } else if (MODE == EPI_BIAS) {
        t += bv;
        if (outF) outF[idx] = t;
        if (outB) outB[idx] = f2bf(t);
      } else { // EPI_SCALE
        t *= scaleIn[idx];
        outB[idx] = f2bf(t);
      }
    }
  }
}

// f32 -> bf16 flat cast
__global__ void cvt_bf16_kernel(const float* __restrict__ in,
                                __bf16* __restrict__ out, int n) {
  int i = blockIdx.x * blockDim.x + threadIdx.x;
  if (i < n) out[i] = f2bf(in[i]);
}

// out[c, r] = bf16(in[r, c]);  in is [R, C] row-major f32.
__global__ void transpose_bf16_kernel(const float* __restrict__ in,
                                      __bf16* __restrict__ out, int R, int C) {
  int i = blockIdx.x * blockDim.x + threadIdx.x;
  if (i < R * C) {
    int r = i / C, c = i - r * C;
    out[(size_t)c * R + r] = f2bf(in[i]);
  }
}

// Per-batch-row reduction + masked output.  One block per row, 256 = N threads.
__global__ __launch_bounds__(256)
void finalize_kernel(const float* __restrict__ fF, const float* __restrict__ kF,
                     const float* __restrict__ GF, const float* __restrict__ JfF,
                     const float* __restrict__ JGF, float* __restrict__ out) {
  const int b = blockIdx.x;
  const int j = threadIdx.x;
  const size_t idx = (size_t)b * N_DIM + j;

  __shared__ float sh0[256], sh1[256], sh2[256];
  __shared__ float factor;

  const float kv  = kF[idx];
  const float jfv = JfF[idx];
  const float jgv = JGF[idx];
  sh0[j] = kv * kv;          // ||k||^2
  sh1[j] = jfv * jfv;        // ||Jf||^2
  sh2[j] = kv * jgv;         // k . JG
  __syncthreads();
#pragma unroll
  for (int s = 128; s > 0; s >>= 1) {
    if (j < s) { sh0[j] += sh0[j + s]; sh1[j] += sh1[j + s]; sh2[j] += sh2[j + s]; }
    __syncthreads();
  }
  if (j == 0) {
    const float kn2  = sh0[0];
    const float kn   = sqrtf(kn2);
    const float kn10 = kn2 * kn2 * kn2 * kn2 * kn2;   // kn^(2*SIGMA)
    const float kn9  = (kn > 0.0f) ? (kn10 / kn) : 0.0f;
    const float c1 = sqrtf(sh1[0]) - 60.0f * kn9;     // ALPHA
    const float c2 = sh2[0] - 20.0f * kn10;           // BETA
    factor = ((c1 > 1e-8f) || (c2 < -1e-8f)) ? 0.5f : 1.0f;
  }
  __syncthreads();
  out[idx] = (fF[idx] + GF[idx]) * factor;
}

// ---------------------------------------------------------------------------
extern "C" void kernel_launch(void* const* d_in, const int* in_sizes, int n_in,
                              void* d_out, int out_size, void* d_ws, size_t ws_size,
                              hipStream_t stream) {
  (void)in_sizes; (void)n_in; (void)out_size; (void)ws_size;
  const float* x   = (const float*)d_in[1];
  const float* Wf1 = (const float*)d_in[2];
  const float* bf1 = (const float*)d_in[3];
  const float* Wf2 = (const float*)d_in[4];
  const float* bf2 = (const float*)d_in[5];
  const float* Wg1 = (const float*)d_in[6];
  const float* bg1 = (const float*)d_in[7];
  const float* Wg2 = (const float*)d_in[8];
  const float* bg2 = (const float*)d_in[9];
  const float* Wk1 = (const float*)d_in[10];
  const float* bk1 = (const float*)d_in[11];
  const float* Wk2 = (const float*)d_in[12];
  const float* bk2 = (const float*)d_in[13];
  float* out = (float*)d_out;

  char* ws = (char*)d_ws;
  size_t off = 0;
  auto alloc = [&](size_t bytes) -> char* {
    off = (off + 255) & ~(size_t)255;
    char* p = ws + off; off += bytes; return p;
  };
  const size_t NW = (size_t)N_DIM * H_DIM;          // 262144
  const size_t BN = (size_t)BS_DIM * N_DIM;         // 32768
  const size_t BH = (size_t)BS_DIM * H_DIM;         // 131072

  __bf16* xb    = (__bf16*)alloc(BN * 2);
  __bf16* Wf1b  = (__bf16*)alloc(NW * 2);
  __bf16* Wg1b  = (__bf16*)alloc(NW * 2);
  __bf16* Wk1b  = (__bf16*)alloc(NW * 2);
  __bf16* Wf2b  = (__bf16*)alloc(NW * 2);
  __bf16* Wg2b  = (__bf16*)alloc(NW * 2);
  __bf16* Wk2b  = (__bf16*)alloc(NW * 2);
  __bf16* Wk1T  = (__bf16*)alloc(NW * 2);   // [256,1024]
  __bf16* Wk2T  = (__bf16*)alloc(NW * 2);   // [1024,256]
  __bf16* hfB   = (__bf16*)alloc(BH * 2);
  __bf16* hgB   = (__bf16*)alloc(BH * 2);
  __bf16* hkB   = (__bf16*)alloc(BH * 2);
  __bf16* fB    = (__bf16*)alloc(BN * 2);
  __bf16* GB    = (__bf16*)alloc(BN * 2);
  __bf16* ufB   = (__bf16*)alloc(BH * 2);
  __bf16* ugB   = (__bf16*)alloc(BH * 2);
  float*  sBuf  = (float*)alloc(BH * 4);
  float*  fF    = (float*)alloc(BN * 4);
  float*  kFv   = (float*)alloc(BN * 4);
  float*  GF    = (float*)alloc(BN * 4);
  float*  JfF   = (float*)alloc(BN * 4);
  float*  JGF   = (float*)alloc(BN * 4);

  auto cvt = [&](const float* in, __bf16* o, int n) {
    cvt_bf16_kernel<<<dim3((n + 255) / 256), dim3(256), 0, stream>>>(in, o, n);
  };
  cvt(x, xb, (int)BN);
  cvt(Wf1, Wf1b, (int)NW);  cvt(Wg1, Wg1b, (int)NW);  cvt(Wk1, Wk1b, (int)NW);
  cvt(Wf2, Wf2b, (int)NW);  cvt(Wg2, Wg2b, (int)NW);  cvt(Wk2, Wk2b, (int)NW);
  transpose_bf16_kernel<<<dim3((int)(NW + 255) / 256), dim3(256), 0, stream>>>(Wk1, Wk1T, H_DIM, N_DIM);
  transpose_bf16_kernel<<<dim3((int)(NW + 255) / 256), dim3(256), 0, stream>>>(Wk2, Wk2T, N_DIM, H_DIM);

#define GEMM(MODE, A, B, BIAS, OUTF, OUTB, SOUT, SCALE, NC, KK)                      \
  gemm_nt_bf16<MODE><<<dim3((NC) / 64), dim3(256), 0, stream>>>(                     \
      A, B, BIAS, OUTF, OUTB, SOUT, SCALE, NC, KK)

  // Layer 1 (tanh, capture sech^2 for k-path)
  GEMM(EPI_TANH, xb, Wf1b, bf1, nullptr, hfB, nullptr, nullptr, H_DIM, N_DIM);
  GEMM(EPI_TANH, xb, Wg1b, bg1, nullptr, hgB, nullptr, nullptr, H_DIM, N_DIM);
  GEMM(EPI_TANH, xb, Wk1b, bk1, nullptr, hkB, sBuf,    nullptr, H_DIM, N_DIM);
  // Layer 2
  GEMM(EPI_BIAS, hfB, Wf2b, bf2, fF,  fB,      nullptr, nullptr, N_DIM, H_DIM);
  GEMM(EPI_BIAS, hgB, Wg2b, bg2, GF,  GB,      nullptr, nullptr, N_DIM, H_DIM);
  GEMM(EPI_BIAS, hkB, Wk2b, bk2, kFv, nullptr, nullptr, nullptr, N_DIM, H_DIM);
  // Jacobian contraction: u = (v @ Wk2) * s ; J*v = u @ Wk1
  GEMM(EPI_SCALE, fB, Wk2T, nullptr, nullptr, ufB, nullptr, sBuf, H_DIM, N_DIM);
  GEMM(EPI_SCALE, GB, Wk2T, nullptr, nullptr, ugB, nullptr, sBuf, H_DIM, N_DIM);
  GEMM(EPI_BIAS, ufB, Wk1T, nullptr, JfF, nullptr, nullptr, nullptr, N_DIM, H_DIM);
  GEMM(EPI_BIAS, ugB, Wk1T, nullptr, JGF, nullptr, nullptr, nullptr, N_DIM, H_DIM);
#undef GEMM

  finalize_kernel<<<dim3(BS_DIM), dim3(N_DIM), 0, stream>>>(fF, kFv, GF, JfF, JGF, out);
}
